// ConformerBlock_21947282882811
// MI455X (gfx1250) — compile-verified, matrix operand order checked by reading, not run on hardware
//
#include <hip/hip_runtime.h>
#include <cstdint>
#include <cstddef>

// ---------------------------------------------------------------------------
// MI455X (gfx1250) Conformer block. All GEMM work on v_wmma_f32_16x16x32_f16,
// fp32 accumulation. wave32. Attention block kept entirely in 320KB WGP LDS.
// Tile staging uses GLOBAL_LOAD_ASYNC_TO_LDS_B128 (ASYNCcnt) when the
// toolchain exposes it; falls back to float4 VGPR staging otherwise.
// ---------------------------------------------------------------------------

typedef __attribute__((ext_vector_type(16))) _Float16 v16h;
typedef __attribute__((ext_vector_type(8)))  _Float16 v8h;
typedef __attribute__((ext_vector_type(8)))  float    v8f;

#define T_TOK   16000      // 8 * 2000 tokens
#define SEQ     2000
#define BATCH   8
#define HID     1024
#define HEADS   8
#define DH      128
#define CS      200        // attention chunk
#define CP      208        // 13*16 : padded rows/cols of score matrix
#define KP      224        // 7*32  : padded K for P@V wmma loop
#define NBLK    10         // 2000 / 200  (rem == 0 -> no mask path needed)
#define ATT_SCALE 0.08838834764831845f   // 128^-0.5

#if defined(__AMDGCN__) && __has_builtin(__builtin_amdgcn_global_load_async_to_lds_b128) && __has_builtin(__builtin_amdgcn_s_wait_asynccnt)
#define USE_ASYNC_LDS 1
#define AS1 __attribute__((address_space(1)))
#define AS3 __attribute__((address_space(3)))
typedef int v4i_g __attribute__((vector_size(16)));   // builtin's pointee type
#else
#define USE_ASYNC_LDS 0
#endif

// --- WMMA fragment loader --------------------------------------------------
// Source stored row-major [row][k] with leading dim `ld` (halves). Per the
// CDNA5 ISA 16-bit A layout: lane<16 -> K 0..7 & 16..23, lane>=16 -> K 8..15 &
// 24..31 (two contiguous 16B runs). B fragments are loaded with the same
// mapping from an N-major ("B^T row-major") buffer (lane -> N row).
__device__ __forceinline__ v16h frag_load(const _Float16* base, int row0, int ld) {
  const int lane = threadIdx.x & 31;
  const _Float16* p = base + (row0 + (lane & 15)) * ld + ((lane >> 4) << 3);
  v8h lo = *(const v8h*)p;
  v8h hi = *(const v8h*)(p + 16);
  v16h out;
#pragma unroll
  for (int i = 0; i < 8; ++i) { out[i] = lo[i]; out[i + 8] = hi[i]; }
  return out;
}

__device__ __forceinline__ v8f wmma_f16(v16h a, v16h b, v8f c) {
  return __builtin_amdgcn_wmma_f32_16x16x32_f16(false, a, false, b, (short)0, c,
                                                false, false);
}

// 16-byte global -> LDS copy, async (ASYNCcnt-tracked) when available.
__device__ __forceinline__ void copy16_g2l(const _Float16* gsrc, _Float16* ldst) {
#if USE_ASYNC_LDS
  __builtin_amdgcn_global_load_async_to_lds_b128((AS1 v4i_g*)gsrc,
                                                 (AS3 v4i_g*)ldst, 0, 0);
#else
  *(float4*)ldst = *(const float4*)gsrc;
#endif
}

__device__ __forceinline__ void async_fence() {
#if USE_ASYNC_LDS
  __builtin_amdgcn_s_wait_asynccnt(0);
#endif
}

// --- Transposed fp32 -> f16 weight conversion ------------------------------
// W [Kd, Nd] row-major fp32  ->  Wt [Nd, Kd] row-major f16. Dims % 32 == 0.
__global__ __launch_bounds__(256) void transpose_f32_f16(
    const float* __restrict__ W, _Float16* __restrict__ Wt, int Kd, int Nd) {
  __shared__ float tile[32][33];
  const int tx = threadIdx.x, ty = threadIdx.y;     // block (32, 8)
  const int nb = blockIdx.x * 32, kb = blockIdx.y * 32;
#pragma unroll
  for (int i = 0; i < 32; i += 8)
    tile[ty + i][tx] = W[(size_t)(kb + ty + i) * Nd + nb + tx];
  __syncthreads();
#pragma unroll
  for (int i = 0; i < 32; i += 8)
    Wt[(size_t)(nb + ty + i) * Kd + kb + tx] = (_Float16)tile[tx][ty + i];
}

// --- rel[c][r][d] = pos_emb[dists[c][r]][d]  (f16) -------------------------
__global__ __launch_bounds__(256) void gather_rel_kernel(
    const int* __restrict__ dists, const float* __restrict__ pos_emb,
    _Float16* __restrict__ rel) {
  const size_t idx = (size_t)blockIdx.x * 256 + threadIdx.x;
  if (idx >= (size_t)CS * CS * DH) return;
  const int d = (int)(idx & (DH - 1));
  const size_t pair = idx >> 7;
  rel[idx] = (_Float16)pos_emb[(size_t)dists[pair] * DH + d];
}

// --- LayerNorm over HID=1024, one block per row ----------------------------
template <typename OutT>
__global__ __launch_bounds__(256) void ln_kernel(
    const float* __restrict__ x, const float* __restrict__ g,
    const float* __restrict__ b, OutT* __restrict__ out) {
  __shared__ float rs[256], rq[256];
  const int tid = threadIdx.x;
  const float* xr = x + (size_t)blockIdx.x * HID;
  float4 xv = *(const float4*)(xr + tid * 4);
  rs[tid] = xv.x + xv.y + xv.z + xv.w;
  rq[tid] = xv.x * xv.x + xv.y * xv.y + xv.z * xv.z + xv.w * xv.w;
  __syncthreads();
  for (int off = 128; off > 0; off >>= 1) {
    if (tid < off) { rs[tid] += rs[tid + off]; rq[tid] += rq[tid + off]; }
    __syncthreads();
  }
  const float mean = rs[0] * (1.0f / HID);
  const float var  = rq[0] * (1.0f / HID) - mean * mean;
  const float rstd = rsqrtf(var + 1e-5f);
  const float4 gv = *(const float4*)(g + tid * 4);
  const float4 bv = *(const float4*)(b + tid * 4);
  OutT* orow = out + (size_t)blockIdx.x * HID;
  orow[tid * 4 + 0] = (OutT)((xv.x - mean) * rstd * gv.x + bv.x);
  orow[tid * 4 + 1] = (OutT)((xv.y - mean) * rstd * gv.y + bv.y);
  orow[tid * 4 + 2] = (OutT)((xv.z - mean) * rstd * gv.z + bv.z);
  orow[tid * 4 + 3] = (OutT)((xv.w - mean) * rstd * gv.w + bv.w);
}

// --- Generic WMMA GEMM: out = epi(A[M,K] @ Bt[N,K]^T + bias) ---------------
// 256 threads = 8 waves; tile 128x128x32; wave -> 64x32 (4x2 accumulators).
// M % 128 == 0, N % 128 == 0, K % 32 == 0 for all call sites.
enum { EPI_PLAIN = 0, EPI_SILU = 1, EPI_GLU = 2, EPI_RESID = 3 };

template <int EPI>
__global__ __launch_bounds__(256) void gemm_f16(
    const _Float16* __restrict__ A, const _Float16* __restrict__ Bt,
    const float* __restrict__ bias, void* __restrict__ out,
    const float* __restrict__ resid, const _Float16* __restrict__ other,
    int M, int N, int K, float scale) {
  __shared__ __align__(16) _Float16 As[128 * 32];
  __shared__ __align__(16) _Float16 Bs[128 * 32];
  const int tid = threadIdx.x;
  const int lane = tid & 31;
  const int wid = tid >> 5;
  const int wm = wid >> 2, wn = wid & 3;
  const int m0 = blockIdx.y * 128;
  const int n0 = blockIdx.x * 128;
  v8f acc[4][2];
#pragma unroll
  for (int i = 0; i < 4; ++i)
#pragma unroll
    for (int j = 0; j < 2; ++j) acc[i][j] = {};

  for (int k0 = 0; k0 < K; k0 += 32) {
#pragma unroll
    for (int i = 0; i < 2; ++i) {
      const int idx = tid * 8 + i * 2048;          // half index into 128x32 tile
      const int row = idx >> 5, col = idx & 31;
      copy16_g2l(&A[(size_t)(m0 + row) * K + k0 + col], &As[idx]);
      copy16_g2l(&Bt[(size_t)(n0 + row) * K + k0 + col], &Bs[idx]);
      if (k0 + 32 < K) {                           // -> global_prefetch_b8
        __builtin_prefetch(&A[(size_t)(m0 + row) * K + k0 + 32 + col], 0, 3);
        __builtin_prefetch(&Bt[(size_t)(n0 + row) * K + k0 + 32 + col], 0, 3);
      }
    }
    async_fence();
    __syncthreads();
    v16h af[4], bf[2];
#pragma unroll
    for (int mi = 0; mi < 4; ++mi) af[mi] = frag_load(As, wm * 64 + mi * 16, 32);
#pragma unroll
    for (int ni = 0; ni < 2; ++ni) bf[ni] = frag_load(Bs, wn * 32 + ni * 16, 32);
#pragma unroll
    for (int mi = 0; mi < 4; ++mi)
#pragma unroll
      for (int ni = 0; ni < 2; ++ni) acc[mi][ni] = wmma_f16(af[mi], bf[ni], acc[mi][ni]);
    __syncthreads();
  }

  // Epilogue. C layout: lane -> n = lane&15, m = vgpr + 8*(lane>>4).
  const int nlane = lane & 15;
  const int mofs = (lane >> 4) << 3;
#pragma unroll
  for (int mi = 0; mi < 4; ++mi)
#pragma unroll
    for (int ni = 0; ni < 2; ++ni) {
      const int gn = n0 + wn * 32 + ni * 16 + nlane;
      const float bv = bias ? bias[gn] : 0.0f;
#pragma unroll
      for (int v = 0; v < 8; ++v) {
        const int gm = m0 + wm * 64 + mi * 16 + mofs + v;
        const size_t o = (size_t)gm * N + gn;
        const float val = acc[mi][ni][v] + bv;
        if constexpr (EPI == EPI_PLAIN) {
          ((_Float16*)out)[o] = (_Float16)val;
        } else if constexpr (EPI == EPI_SILU) {
          ((_Float16*)out)[o] = (_Float16)(val / (1.0f + __expf(-val)));
        } else if constexpr (EPI == EPI_GLU) {
          ((_Float16*)out)[o] =
              (_Float16)((float)other[o] / (1.0f + __expf(-val)));
        } else {  // EPI_RESID
          ((float*)out)[o] = resid[o] + scale * val;
        }
      }
    }
}

// --- Attention: one workgroup per (batch, chunk, head) ---------------------
// LDS: S fp32 [208][208] | region2 = { Q,K f16 [208][128] } then reused as
// { V^T f16 [128][224], P f16 [208][224] }. Total 323,584 B <= 320 KB WGP LDS.
#define SMEM_ATTN (CP * CP * 4 + 128 * KP * 2 + CP * KP * 2)

__global__ __launch_bounds__(256) void attn_kernel(
    const _Float16* __restrict__ q, const _Float16* __restrict__ kv,
    const _Float16* __restrict__ rel, _Float16* __restrict__ octx) {
  extern __shared__ char smem[];
  float* S = (float*)smem;
  _Float16* qs = (_Float16*)(smem + CP * CP * 4);
  _Float16* ks = qs + CP * 128;
  _Float16* vT = (_Float16*)(smem + CP * CP * 4);   // reuses q/k region
  _Float16* Ps = vT + 128 * KP;

  const int tid = threadIdx.x;
  const int lane = tid & 31;
  const int wid = tid >> 5;
  const int b = blockIdx.x / (NBLK * HEADS);
  const int m = (blockIdx.x / HEADS) % NBLK;
  const int h = blockIdx.x % HEADS;
  const size_t t0 = (size_t)b * SEQ + (size_t)m * CS;
  const float4 z4 = {0.f, 0.f, 0.f, 0.f};

  // phase 1: stage Q, K (async global->LDS; zero-pad rows 200..207)
  for (int i = tid; i < CP * 16; i += 256) {
    const int row = i >> 4;
    const int ch = (i & 15) << 3;
    if (row < CS) {
      copy16_g2l(&q[(t0 + row) * HID + h * DH + ch], &qs[row * 128 + ch]);
      copy16_g2l(&kv[(t0 + row) * (2 * HID) + h * DH + ch], &ks[row * 128 + ch]);
    } else {
      *(float4*)&qs[row * 128 + ch] = z4;
      *(float4*)&ks[row * 128 + ch] = z4;
    }
  }
  async_fence();
  __syncthreads();

  // phase 2: S = (Q K^T + Q.rel) * scale ; WMMA for QK^T, VALU dot for rel
  for (int tile = wid; tile < 13 * 13; tile += 8) {
    const int tm = tile / 13, tn = tile % 13;
    v8f acc = {};
#pragma unroll
    for (int kk = 0; kk < 4; ++kk)
      acc = wmma_f16(frag_load(qs + kk * 32, tm * 16, 128),
                     frag_load(ks + kk * 32, tn * 16, 128), acc);
    const int nlane = lane & 15;
    const int mofs = (lane >> 4) << 3;
#pragma unroll
    for (int v = 0; v < 8; ++v) {
      const int c = tm * 16 + mofs + v;
      const int r = tn * 16 + nlane;
      float val = acc[v];
      if (c < CS && r < CS) {
        const _Float16* rp = rel + ((size_t)c * CS + r) * DH;
        const _Float16* qp = qs + c * 128;
        float pos = 0.f;
#pragma unroll 4
        for (int dch = 0; dch < 16; ++dch) {
          v8h qv = *(const v8h*)(qp + dch * 8);
          v8h rv = *(const v8h*)(rp + dch * 8);
#pragma unroll
          for (int j = 0; j < 8; ++j) pos += (float)qv[j] * (float)rv[j];
        }
        val += pos;
      }
      S[c * CP + r] = val * ATT_SCALE;
    }
  }
  __syncthreads();   // q/k reads done; region2 may be reused

  // phase 3: zero V^T, stage V^T, softmax rows -> P (f16, K padded to 224)
  for (int i = tid; i < (128 * KP) / 8; i += 256) ((float4*)vT)[i] = z4;
  __syncthreads();
  for (int i = tid; i < CS * 16; i += 256) {
    const int r = i >> 4;
    const int ch = (i & 15) << 3;
    v8h vv = *(const v8h*)&kv[(t0 + r) * (2 * HID) + HID + h * DH + ch];
#pragma unroll
    for (int j = 0; j < 8; ++j) vT[(ch + j) * KP + r] = vv[j];
  }
  if (tid < CP) {
    const int c = tid;
    float* Sr = S + c * CP;
    _Float16* Pr = Ps + c * KP;
    if (c < CS) {
      float mx = -3.4e38f;
      for (int r = 0; r < CS; ++r) mx = fmaxf(mx, Sr[r]);
      float sum = 0.f;
      for (int r = 0; r < CS; ++r) { float e = __expf(Sr[r] - mx); Sr[r] = e; sum += e; }
      const float inv = 1.0f / sum;
      for (int r = 0; r < CS; ++r) Pr[r] = (_Float16)(Sr[r] * inv);
      for (int r = CS; r < KP; ++r) Pr[r] = (_Float16)0.f;
    } else {
      for (int r = 0; r < KP; ++r) Pr[r] = (_Float16)0.f;
    }
  }
  __syncthreads();

  // phase 4: O = P @ V  (13x8 output tiles, K = 224)
  for (int tile = wid; tile < 13 * 8; tile += 8) {
    const int tm = tile >> 3;
    const int tn = tile & 7;
    v8f acc = {};
#pragma unroll
    for (int kk = 0; kk < 7; ++kk)
      acc = wmma_f16(frag_load(Ps + kk * 32, tm * 16, KP),
                     frag_load(vT + kk * 32, tn * 16, KP), acc);
    const int nlane = lane & 15;
    const int mofs = (lane >> 4) << 3;
#pragma unroll
    for (int v = 0; v < 8; ++v) {
      const int c = tm * 16 + mofs + v;
      const int d = tn * 16 + nlane;
      if (c < CS) octx[(t0 + c) * HID + h * DH + d] = (_Float16)acc[v];
    }
  }
}

// --- depthwise conv (K=15, pad 7) + BatchNorm(eval) + SiLU -----------------
__global__ __launch_bounds__(256) void dwconv_bn_silu(
    const _Float16* __restrict__ in, const float* __restrict__ w,
    const float* __restrict__ g, const float* __restrict__ bb,
    const float* __restrict__ mn, const float* __restrict__ vr,
    _Float16* __restrict__ out) {
  const size_t idx = (size_t)blockIdx.x * 256 + threadIdx.x;
  const size_t total = (size_t)T_TOK * 2048;
  if (idx >= total) return;
  const int c = (int)(idx & 2047);
  const size_t tok = idx >> 11;             // b*2000 + n
  const int n = (int)(tok % SEQ);
  const size_t brow = tok - n;              // b*2000
  float acc = 0.f;
  const float* wc = w + c * 15;
#pragma unroll
  for (int k = 0; k < 15; ++k) {
    const int nn = n + k - 7;
    if (nn >= 0 && nn < SEQ)
      acc += (float)in[((brow + nn) << 11) + c] * wc[k];
  }
  const float hb = (acc - mn[c]) * rsqrtf(vr[c] + 1e-5f) * g[c] + bb[c];
  out[idx] = (_Float16)(hb / (1.0f + __expf(-hb)));   // silu
}

// ---------------------------------------------------------------------------
extern "C" void kernel_launch(void* const* d_in, const int* in_sizes, int n_in,
                              void* d_out, int out_size, void* d_ws, size_t ws_size,
                              hipStream_t stream) {
  (void)in_sizes; (void)n_in; (void)out_size; (void)ws_size;
  const float* x       = (const float*)d_in[0];
  const int*   dists   = (const int*)d_in[1];
  const float* ff1_ng  = (const float*)d_in[2];
  const float* ff1_nb  = (const float*)d_in[3];
  const float* ff1_w1  = (const float*)d_in[4];
  const float* ff1_b1  = (const float*)d_in[5];
  const float* ff1_w2  = (const float*)d_in[6];
  const float* ff1_b2  = (const float*)d_in[7];
  const float* attn_ng = (const float*)d_in[8];
  const float* attn_nb = (const float*)d_in[9];
  const float* wq      = (const float*)d_in[10];
  const float* wkv     = (const float*)d_in[11];
  const float* pos_emb = (const float*)d_in[12];
  const float* wo      = (const float*)d_in[13];
  const float* bo      = (const float*)d_in[14];
  const float* conv_ng = (const float*)d_in[15];
  const float* conv_nb = (const float*)d_in[16];
  const float* pw1_w   = (const float*)d_in[17];
  const float* pw1_b   = (const float*)d_in[18];
  const float* dw_w    = (const float*)d_in[19];
  const float* bn_g    = (const float*)d_in[20];
  const float* bn_b    = (const float*)d_in[21];
  const float* bn_m    = (const float*)d_in[22];
  const float* bn_v    = (const float*)d_in[23];
  const float* pw2_w   = (const float*)d_in[24];
  const float* pw2_b   = (const float*)d_in[25];
  const float* ff2_ng  = (const float*)d_in[26];
  const float* ff2_nb  = (const float*)d_in[27];
  const float* ff2_w1  = (const float*)d_in[28];
  const float* ff2_b1  = (const float*)d_in[29];
  const float* ff2_w2  = (const float*)d_in[30];
  const float* ff2_b2  = (const float*)d_in[31];
  const float* post_g  = (const float*)d_in[32];
  const float* post_b  = (const float*)d_in[33];

  // --- workspace bump allocator (total ~524 MB) ---
  char* ws = (char*)d_ws;
  size_t off = 0;
  auto alloc = [&](size_t bytes) -> char* {
    char* p = ws + off;
    off += (bytes + 255) & ~(size_t)255;
    return p;
  };
  const size_t T = T_TOK;
  _Float16* w11t = (_Float16*)alloc((size_t)4096 * 1024 * 2);
  _Float16* w12t = (_Float16*)alloc((size_t)1024 * 4096 * 2);
  _Float16* wqt  = (_Float16*)alloc((size_t)1024 * 1024 * 2);
  _Float16* wkvt = (_Float16*)alloc((size_t)2048 * 1024 * 2);
  _Float16* wot  = (_Float16*)alloc((size_t)1024 * 1024 * 2);
  _Float16* pw1t = (_Float16*)alloc((size_t)4096 * 1024 * 2);
  _Float16* pw2t = (_Float16*)alloc((size_t)1024 * 2048 * 2);
  _Float16* w21t = (_Float16*)alloc((size_t)4096 * 1024 * 2);
  _Float16* w22t = (_Float16*)alloc((size_t)1024 * 4096 * 2);
  _Float16* relh = (_Float16*)alloc((size_t)CS * CS * DH * 2);
  _Float16* lnh  = (_Float16*)alloc(T * HID * 2);      // LN out / attn context
  _Float16* hbig = (_Float16*)alloc(T * 4096 * 2);     // FF mid / KV / GLU a-half
  _Float16* qbuf = (_Float16*)alloc(T * HID * 2);
  _Float16* glub = (_Float16*)alloc(T * 2048 * 2);
  _Float16* dwob = (_Float16*)alloc(T * 2048 * 2);
  float*    xa   = (float*)alloc(T * HID * 4);
  float*    xb   = (float*)alloc(T * HID * 4);

  const dim3 tb(32, 8);
  // Weight transpose+convert (W[K,N] -> Wt[N,K] f16); grid (N/32, K/32)
  transpose_f32_f16<<<dim3(4096/32, 1024/32), tb, 0, stream>>>(ff1_w1, w11t, 1024, 4096);
  transpose_f32_f16<<<dim3(1024/32, 4096/32), tb, 0, stream>>>(ff1_w2, w12t, 4096, 1024);
  transpose_f32_f16<<<dim3(1024/32, 1024/32), tb, 0, stream>>>(wq,     wqt,  1024, 1024);
  transpose_f32_f16<<<dim3(2048/32, 1024/32), tb, 0, stream>>>(wkv,    wkvt, 1024, 2048);
  transpose_f32_f16<<<dim3(1024/32, 1024/32), tb, 0, stream>>>(wo,     wot,  1024, 1024);
  transpose_f32_f16<<<dim3(4096/32, 1024/32), tb, 0, stream>>>(pw1_w,  pw1t, 1024, 4096);
  transpose_f32_f16<<<dim3(1024/32, 2048/32), tb, 0, stream>>>(pw2_w,  pw2t, 2048, 1024);
  transpose_f32_f16<<<dim3(4096/32, 1024/32), tb, 0, stream>>>(ff2_w1, w21t, 1024, 4096);
  transpose_f32_f16<<<dim3(1024/32, 4096/32), tb, 0, stream>>>(ff2_w2, w22t, 4096, 1024);
  gather_rel_kernel<<<(CS * CS * DH + 255) / 256, 256, 0, stream>>>(dists, pos_emb, relh);

  const int GY = T_TOK / 128;   // 125
  // ---- FF1 (macaron, 0.5 residual) ----
  ln_kernel<_Float16><<<T_TOK, 256, 0, stream>>>(x, ff1_ng, ff1_nb, lnh);
  gemm_f16<EPI_SILU><<<dim3(4096/128, GY), 256, 0, stream>>>(
      lnh, w11t, ff1_b1, hbig, nullptr, nullptr, T_TOK, 4096, 1024, 1.f);
  gemm_f16<EPI_RESID><<<dim3(1024/128, GY), 256, 0, stream>>>(
      hbig, w12t, ff1_b2, xa, x, nullptr, T_TOK, 1024, 4096, 0.5f);

  // ---- Attention ----
  ln_kernel<_Float16><<<T_TOK, 256, 0, stream>>>(xa, attn_ng, attn_nb, lnh);
  gemm_f16<EPI_PLAIN><<<dim3(1024/128, GY), 256, 0, stream>>>(
      lnh, wqt, nullptr, qbuf, nullptr, nullptr, T_TOK, 1024, 1024, 1.f);
  gemm_f16<EPI_PLAIN><<<dim3(2048/128, GY), 256, 0, stream>>>(
      lnh, wkvt, nullptr, hbig, nullptr, nullptr, T_TOK, 2048, 1024, 1.f);
  attn_kernel<<<BATCH * NBLK * HEADS, 256, SMEM_ATTN, stream>>>(
      qbuf, hbig, relh, lnh);   // context written into lnh
  gemm_f16<EPI_RESID><<<dim3(1024/128, GY), 256, 0, stream>>>(
      lnh, wot, bo, xb, xa, nullptr, T_TOK, 1024, 1024, 1.f);

  // ---- Conv module ----
  ln_kernel<_Float16><<<T_TOK, 256, 0, stream>>>(xb, conv_ng, conv_nb, lnh);
  // pw1: a-half plain, gate-half fused GLU combine (reads a from hbig)
  gemm_f16<EPI_PLAIN><<<dim3(2048/128, GY), 256, 0, stream>>>(
      lnh, pw1t, pw1_b, hbig, nullptr, nullptr, T_TOK, 2048, 1024, 1.f);
  gemm_f16<EPI_GLU><<<dim3(2048/128, GY), 256, 0, stream>>>(
      lnh, pw1t + (size_t)2048 * 1024, pw1_b + 2048, glub, nullptr, hbig,
      T_TOK, 2048, 1024, 1.f);
  dwconv_bn_silu<<<(int)(((size_t)T_TOK * 2048 + 255) / 256), 256, 0, stream>>>(
      glub, dw_w, bn_g, bn_b, bn_m, bn_v, dwob);
  gemm_f16<EPI_RESID><<<dim3(1024/128, GY), 256, 0, stream>>>(
      dwob, pw2t, pw2_b, xa, xb, nullptr, T_TOK, 1024, 2048, 1.f);

  // ---- FF2 (macaron, 0.5 residual) ----
  ln_kernel<_Float16><<<T_TOK, 256, 0, stream>>>(xa, ff2_ng, ff2_nb, lnh);
  gemm_f16<EPI_SILU><<<dim3(4096/128, GY), 256, 0, stream>>>(
      lnh, w21t, ff2_b1, hbig, nullptr, nullptr, T_TOK, 4096, 1024, 1.f);
  gemm_f16<EPI_RESID><<<dim3(1024/128, GY), 256, 0, stream>>>(
      hbig, w22t, ff2_b2, xb, xa, nullptr, T_TOK, 1024, 4096, 0.5f);

  // ---- final LayerNorm -> d_out (fp32) ----
  ln_kernel<float><<<T_TOK, 256, 0, stream>>>(xb, post_g, post_b, (float*)d_out);
}
